// Net_86114094284913
// MI455X (gfx1250) — matrix operand order repaired and checked by reading, not compiled, not run on hardware
//
#include <hip/hip_runtime.h>

typedef __attribute__((ext_vector_type(16))) _Float16 v16h;
typedef __attribute__((ext_vector_type(8)))  float    v8f;

#define DD 64

__device__ __forceinline__ v8f wmma16(v16h a, v16h b, v8f c) {
  // v_wmma_f32_16x16x32_f16 : D = A(16x32 f16) * B(32x16 f16) + C(16x16 f32)
  return __builtin_amdgcn_wmma_f32_16x16x32_f16(
      /*neg_a=*/false, a, /*neg_b=*/false, b,
      /*c_mod=*/(short)0, c, /*reuse_a=*/false, /*reuse_b=*/false);
}

// A fragment: A[m,k] = X[tile+m, kbase+k], X row-major f32 with ld=64.
// ISA 16-bit A 16x32 layout: lanes 0-15 -> M=lane, K = {0..7, 16..23};
//                            lanes 16-31 -> M=lane-16, K = {8..15, 24..31}.
__device__ __forceinline__ v16h load_A_f32(const float* X, int tile, int kbase,
                                           int lane, int maxRow) {
  int m = tile + (lane & 15);
  if (m > maxRow) m = maxRow;
  const float* p = X + (long long)m * DD;
  int k0 = kbase + ((lane & 16) ? 8 : 0);
  v16h a;
#pragma unroll
  for (int i = 0; i < 8; ++i) a[i] = (_Float16)p[k0 + i];
#pragma unroll
  for (int i = 0; i < 8; ++i) a[8 + i] = (_Float16)p[k0 + 16 + i];
  return a;
}

// B fragment for X @ W^T : B[k,n] = W[n,k], W row-major [64,64].
// 16-bit B 32x16 layout: lanes 0-15 -> N=lane, K = kbase+{0..7, 16..23};
//                        lanes 16-31 -> N=lane-16, K = kbase+{8..15, 24..31}.
__device__ __forceinline__ v16h load_B_WT(const float* W, int nbase, int kbase,
                                          int lane) {
  int n = nbase + (lane & 15);
  const float* p = W + n * DD + kbase + ((lane & 16) ? 8 : 0);
  v16h b;
#pragma unroll
  for (int i = 0; i < 16; ++i) b[i] = (_Float16)p[(i < 8) ? i : (i + 8)];
  return b;
}

// ---------------------------------------------------------------- utilities
__global__ void zero_kernel(float* p, size_t n) {
  size_t i = (size_t)blockIdx.x * blockDim.x + threadIdx.x;
  size_t stride = (size_t)gridDim.x * blockDim.x;
  for (; i < n; i += stride) p[i] = 0.0f;
}

// ----------------------------------------- edge pass 1: h_re sum, deg, S[n,5]
__global__ void edge_pass1(const float* __restrict__ review_feat,
                           const int* __restrict__ src,
                           const int* __restrict__ dst,
                           const int* __restrict__ score,
                           const float* __restrict__ ci,
                           float* __restrict__ h_re, float* __restrict__ deg,
                           float* __restrict__ S, long long E64) {
  long long t = (long long)blockIdx.x * blockDim.x + threadIdx.x;
  if (t >= E64) return;
  int e = (int)(t >> 6);
  int d = (int)(t & 63);
  int dn = dst[e];
  atomicAdd(&h_re[(long long)dn * DD + d], review_feat[t]);
  if (d == 0) {
    atomicAdd(&deg[dn], 1.0f);
    atomicAdd(&S[dn * 5 + score[e]], ci[src[e]]);
  }
}

__global__ void hre_div(float* __restrict__ h_re, const float* __restrict__ deg,
                        long long N64) {
  long long t = (long long)blockIdx.x * blockDim.x + threadIdx.x;
  if (t >= N64) return;
  int n = (int)(t >> 6);
  h_re[t] /= fmaxf(deg[n], 1.0f);
}

// ------------------------------------------- column sums for the mean vectors
__global__ void colsum_kernel(const float* __restrict__ X,
                              float* __restrict__ out64, int N,
                              int rows_per_block) {
  int d = threadIdx.x;  // 64 threads
  int r0 = blockIdx.x * rows_per_block;
  int r1 = r0 + rows_per_block;
  if (r1 > N) r1 = N;
  float acc = 0.0f;
  for (int r = r0; r < r1; ++r) acc += X[(long long)r * DD + d];
  atomicAdd(&out64[d], acc);
}

// -------------------- WMMA node GEMM: r_fe3 = h_re@W1^T ; A2 = f2 + h_re@W2^T
__global__ void node_gemm(const float* __restrict__ h_re,
                          const float* __restrict__ feature2,
                          const float* __restrict__ W_r1,
                          const float* __restrict__ W_r2,
                          float* __restrict__ r_fe3, float* __restrict__ A2,
                          int N) {
  int tile = blockIdx.x * 16;
  if (tile >= N) return;
  int lane = threadIdx.x;
  v16h a0 = load_A_f32(h_re, tile, 0, lane, N - 1);
  v16h a1 = load_A_f32(h_re, tile, 32, lane, N - 1);
#pragma unroll
  for (int nt = 0; nt < 4; ++nt) {
    int nb = nt * 16;
    v8f c1 = {};
    v8f c2 = {};
    v16h b = load_B_WT(W_r1, nb, 0, lane);
    c1 = wmma16(a0, b, c1);
    b = load_B_WT(W_r1, nb, 32, lane);
    c1 = wmma16(a1, b, c1);
    b = load_B_WT(W_r2, nb, 0, lane);
    c2 = wmma16(a0, b, c2);
    b = load_B_WT(W_r2, nb, 32, lane);
    c2 = wmma16(a1, b, c2);
    // C/D layout: VGPR i, lanes 0-15 -> M=i, lanes 16-31 -> M=8+i; N=lane%16
    int col = nb + (lane & 15);
    int ro = (lane & 16) ? 8 : 0;
#pragma unroll
    for (int i = 0; i < 8; ++i) {
      int row = tile + ro + i;
      if (row < N) {
        long long off = (long long)row * DD + col;
        r_fe3[off] = c1[i];
        A2[off] = feature2[off] + c2[i];
      }
    }
  }
}

// ------------------------------- tiny kernel: mean vectors m1,m2,m3 (64 each)
__global__ void means_kernel(const float* __restrict__ hsum,
                             const float* __restrict__ f2sum,
                             const float* __restrict__ f3sum,
                             const float* __restrict__ W_r1,
                             const float* __restrict__ W_r2,
                             float* __restrict__ m1, float* __restrict__ m2,
                             float* __restrict__ m3, float invN) {
  __shared__ float hm[DD];
  int d = threadIdx.x;  // 64 threads
  hm[d] = hsum[d] * invN;
  __syncthreads();
  float a1 = 0.0f, a2 = 0.0f;
#pragma unroll
  for (int k = 0; k < DD; ++k) {
    a1 += hm[k] * W_r1[d * DD + k];
    a2 += hm[k] * W_r2[d * DD + k];
  }
  m1[d] = a1;                   // rf3_mean
  m2[d] = f2sum[d] * invN + a2; // f2_mean + rf2_mean
  m3[d] = f3sum[d] * invN;      // f3_mean
}

// ----------------------- edge pass 2: three gather->scale->scatter reductions
__global__ void edge_pass2(const int* __restrict__ src,
                           const int* __restrict__ dst,
                           const int* __restrict__ score,
                           const float* __restrict__ ci,
                           const float* __restrict__ r_fe3,
                           const float* __restrict__ A2,
                           const float* __restrict__ feature3,
                           const float* __restrict__ E1,
                           const float* __restrict__ E2,
                           const float* __restrict__ E3,
                           float* __restrict__ rst_re, float* __restrict__ rst,
                           float* __restrict__ rst_id, long long E64) {
  long long t = (long long)blockIdx.x * blockDim.x + threadIdx.x;
  if (t >= E64) return;
  int e = (int)(t >> 6);
  int d = (int)(t & 63);
  int s = src[e];
  int dn = dst[e];
  int c = score[e];
  float cis = ci[s];
  long long so = (long long)s * DD + d;
  long long doff = (long long)dn * DD + d;
  atomicAdd(&rst_re[doff], r_fe3[so] * E1[c * DD + d] * cis);
  atomicAdd(&rst[doff], A2[so] * E2[c * DD + d] * cis);
  atomicAdd(&rst_id[doff], feature3[so] * E3[c * DD + d] * cis);
}

// --------------- finalize: scale by ci; reconstruct the three freeze outputs
__global__ void finalize_nodes(const float* __restrict__ ci,
                               const float* __restrict__ S,
                               const float* __restrict__ m1,
                               const float* __restrict__ m2,
                               const float* __restrict__ m3,
                               const float* __restrict__ E1,
                               const float* __restrict__ E2,
                               const float* __restrict__ E3,
                               float* __restrict__ rst,
                               float* __restrict__ rst_freeze,
                               float* __restrict__ rst_re,
                               float* __restrict__ rst_id,
                               float* __restrict__ rst_re_freeze,
                               float* __restrict__ rst_id_freeze,
                               long long N64) {
  long long t = (long long)blockIdx.x * blockDim.x + threadIdx.x;
  if (t >= N64) return;
  int n = (int)(t >> 6);
  int d = (int)(t & 63);
  float cin = ci[n];
  rst[t] *= cin;
  rst_re[t] *= cin;
  rst_id[t] *= cin;
  float a1 = 0.0f, a2 = 0.0f, a3 = 0.0f;
#pragma unroll
  for (int c = 0; c < 5; ++c) {
    float s = S[n * 5 + c];
    a1 += s * E1[c * DD + d];
    a2 += s * E2[c * DD + d];
    a3 += s * E3[c * DD + d];
  }
  rst_re_freeze[t] = cin * m1[d] * a1;
  rst_freeze[t] = cin * m2[d] * a2;
  rst_id_freeze[t] = cin * m3[d] * a3;
}

// --------------------------- WMMA predictor head: 16-row tiles, both variants
__global__ void predict_kernel(const float* __restrict__ rst,
                               const float* __restrict__ rst_freeze,
                               const int* __restrict__ users,
                               const int* __restrict__ items,
                               const float* __restrict__ P1W,
                               const float* __restrict__ P1b,
                               const float* __restrict__ P2W,
                               const float* __restrict__ P2b,
                               float* __restrict__ pred,
                               float* __restrict__ pred_freeze, int B) {
  __shared__ float hbuf[16 * DD];
  int tile = blockIdx.x * 16;
  if (tile >= B) return;
  int lane = threadIdx.x;
  int maxRow = B - 1;
  for (int variant = 0; variant < 2; ++variant) {
    const float* isrc = variant ? rst_freeze : rst;
    float* out = variant ? pred_freeze : pred;
    // Build A = (ue * ie) in f16 with the 16-bit A layout.
    int row = tile + (lane & 15);
    if (row > maxRow) row = maxRow;
    const float* up = rst + (long long)users[row] * DD;
    const float* ip = isrc + (long long)items[row] * DD;
    int k0 = (lane & 16) ? 8 : 0;
    v16h a0, a1;
#pragma unroll
    for (int i = 0; i < 8; ++i) {
      a0[i] = (_Float16)(up[k0 + i] * ip[k0 + i]);
      a0[8 + i] = (_Float16)(up[k0 + 16 + i] * ip[k0 + 16 + i]);
      a1[i] = (_Float16)(up[32 + k0 + i] * ip[32 + k0 + i]);
      a1[8 + i] = (_Float16)(up[32 + k0 + 16 + i] * ip[32 + k0 + 16 + i]);
    }
#pragma unroll
    for (int nt = 0; nt < 4; ++nt) {
      int nb = nt * 16;
      v8f c = {};
      v16h b = load_B_WT(P1W, nb, 0, lane);
      c = wmma16(a0, b, c);
      b = load_B_WT(P1W, nb, 32, lane);
      c = wmma16(a1, b, c);
      int col = nb + (lane & 15);
      int ro = (lane & 16) ? 8 : 0;
#pragma unroll
      for (int i = 0; i < 8; ++i) {
        float v = c[i] + P1b[col];
        v = v > 0.0f ? v : 0.1f * v;  // LeakyReLU(0.1)
        hbuf[(ro + i) * DD + col] = v;
      }
    }
    __syncthreads();
    // Second layer: 16 rows x 5 outputs, 80 dot products of length 64.
    for (int idx = lane; idx < 80; idx += 32) {
      int r = idx / 5, o = idx % 5;
      float acc = P2b[o];
      const float* hr = &hbuf[r * DD];
      const float* w = &P2W[o * DD];
#pragma unroll
      for (int k = 0; k < DD; ++k) acc += hr[k] * w[k];
      int orow = tile + r;
      if (orow < B) out[(long long)orow * 5 + o] = acc;
    }
    __syncthreads();
  }
}

// ---------------------------------------------------------------------------
extern "C" void kernel_launch(void* const* d_in, const int* in_sizes, int n_in,
                              void* d_out, int out_size, void* d_ws,
                              size_t ws_size, hipStream_t stream) {
  const float* review_feat = (const float*)d_in[0];
  const float* ci = (const float*)d_in[1];
  const int* src = (const int*)d_in[2];
  const int* dst = (const int*)d_in[3];
  const int* score = (const int*)d_in[4];
  const int* users = (const int*)d_in[5];
  const int* items = (const int*)d_in[6];
  const float* W_r1 = (const float*)d_in[7];
  const float* W_r2 = (const float*)d_in[8];
  const float* feature2 = (const float*)d_in[9];
  const float* feature3 = (const float*)d_in[10];
  const float* E1 = (const float*)d_in[11];
  const float* E2 = (const float*)d_in[12];
  const float* E3 = (const float*)d_in[13];
  const float* P1W = (const float*)d_in[14];
  const float* P1b = (const float*)d_in[15];
  const float* P2W = (const float*)d_in[16];
  const float* P2b = (const float*)d_in[17];

  const int N = in_sizes[1];       // 200000
  const int E = in_sizes[2];       // 1000000
  const int B = in_sizes[5];       // 8192
  const long long N64 = (long long)N * DD;
  const long long E64 = (long long)E * DD;

  // Workspace layout (floats): h_re | deg | S(5N) | hsum | f2sum | f3sum |
  //                            m1 | m2 | m3 | r_fe3 | A2
  float* ws = (float*)d_ws;
  float* h_re = ws;
  float* deg = h_re + N64;
  float* S = deg + N;
  float* hsum = S + (long long)5 * N;
  float* f2sum = hsum + DD;
  float* f3sum = f2sum + DD;
  float* m1 = f3sum + DD;
  float* m2 = m1 + DD;
  float* m3 = m2 + DD;
  float* r_fe3 = m3 + DD;
  float* A2 = r_fe3 + N64;

  // Output layout (floats): pred(B*5) | pred_freeze(B*5) | rst | rst_freeze |
  //                         rst_re | rst_id | rst_re_freeze | rst_id_freeze
  float* out = (float*)d_out;
  float* pred = out;
  float* pred_freeze = pred + (long long)B * 5;
  float* rst = pred_freeze + (long long)B * 5;
  float* rst_freeze = rst + N64;
  float* rst_re = rst_freeze + N64;
  float* rst_id = rst_re + N64;
  float* rst_re_freeze = rst_id + N64;
  float* rst_id_freeze = rst_re_freeze + N64;

  // 1) zero accumulators (ws front region is contiguous; rst_re/rst_id too)
  size_t zn0 = (size_t)(N64 + 6LL * N + 3 * DD);
  zero_kernel<<<4096, 256, 0, stream>>>(h_re, zn0);
  zero_kernel<<<4096, 256, 0, stream>>>(rst, (size_t)N64);
  zero_kernel<<<4096, 256, 0, stream>>>(rst_re, (size_t)(2 * N64));

  // 2) edge pass 1: h_re sums, degrees, ci-weighted score-class table S
  {
    long long nthr = E64;
    int blocks = (int)((nthr + 255) / 256);
    edge_pass1<<<blocks, 256, 0, stream>>>(review_feat, src, dst, score, ci,
                                           h_re, deg, S, E64);
  }

  // 3) h_re /= max(deg,1)
  {
    int blocks = (int)((N64 + 255) / 256);
    hre_div<<<blocks, 256, 0, stream>>>(h_re, deg, N64);
  }

  // 4) column sums for mean vectors
  {
    int rpb = 512;
    int blocks = (N + rpb - 1) / rpb;
    colsum_kernel<<<blocks, 64, 0, stream>>>(h_re, hsum, N, rpb);
    colsum_kernel<<<blocks, 64, 0, stream>>>(feature2, f2sum, N, rpb);
    colsum_kernel<<<blocks, 64, 0, stream>>>(feature3, f3sum, N, rpb);
  }

  // 5) node GEMMs via WMMA: r_fe3 = h_re@W_r1^T ; A2 = feature2 + h_re@W_r2^T
  {
    int blocks = (N + 15) / 16;
    node_gemm<<<blocks, 32, 0, stream>>>(h_re, feature2, W_r1, W_r2, r_fe3, A2,
                                         N);
  }

  // 6) mean vectors
  means_kernel<<<1, 64, 0, stream>>>(hsum, f2sum, f3sum, W_r1, W_r2, m1, m2,
                                     m3, 1.0f / (float)N);

  // 7) edge pass 2: gather/scale/scatter for rst_re, rst, rst_id accumulators
  {
    int blocks = (int)((E64 + 255) / 256);
    edge_pass2<<<blocks, 256, 0, stream>>>(src, dst, score, ci, r_fe3, A2,
                                           feature3, E1, E2, E3, rst_re, rst,
                                           rst_id, E64);
  }

  // 8) finalize node outputs (scale by ci, reconstruct freeze outputs)
  {
    int blocks = (int)((N64 + 255) / 256);
    finalize_nodes<<<blocks, 256, 0, stream>>>(ci, S, m1, m2, m3, E1, E2, E3,
                                               rst, rst_freeze, rst_re, rst_id,
                                               rst_re_freeze, rst_id_freeze,
                                               N64);
  }

  // 9) predictor head via WMMA
  {
    int blocks = (B + 15) / 16;
    predict_kernel<<<blocks, 32, 0, stream>>>(rst, rst_freeze, users, items,
                                              P1W, P1b, P2W, P2b, pred,
                                              pred_freeze, B);
  }
}
